// MultiScaleDeformableAttention_67765993997298
// MI455X (gfx1250) — matrix operand order repaired and checked by reading, not compiled, not run on hardware
//
#include <hip/hip_runtime.h>
#include <hip/hip_bf16.h>

typedef __bf16 v16bf __attribute__((ext_vector_type(16)));
typedef float  v8f   __attribute__((ext_vector_type(8)));

// Problem constants (match reference)
static constexpr int kE  = 256;
static constexpr int kM  = 8;
static constexpr int kL  = 4;
static constexpr int kP  = 4;
static constexpr int kD  = 32;          // E / M
static constexpr int kB  = 2;
static constexpr int kNV = 19560;       // sum of level H*W
static constexpr int kLQ = 19560;
static constexpr int kROWS = kB * kLQ;  // 39120 rows for all GEMMs (divisible by 16)

__device__ __constant__ int cLVL_H[4] = {92, 46, 23, 12};
__device__ __constant__ int cLVL_W[4] = {160, 80, 40, 20};
__device__ __constant__ int cLVL_S[4] = {0, 14720, 18400, 19320};

// ---------------------------------------------------------------------------
// Pack f32 weight [K][N] (row-major) -> bf16 [N][K] (transposed) so that a
// lane's B fragment (16 consecutive K values for one column) is one
// contiguous 32-byte load.
// ---------------------------------------------------------------------------
__global__ void pack_w_bf16(const float* __restrict__ W, __bf16* __restrict__ Wt,
                            int N, int K) {
    int tid = blockIdx.x * blockDim.x + threadIdx.x;
    if (tid >= N * K) return;
    int n = tid / K;
    int k = tid - n * K;
    Wt[(size_t)n * K + k] = (__bf16)W[(size_t)k * N + n];
}

// ---------------------------------------------------------------------------
// WMMA GEMM: C[M,N] = A(f32,row-major)[M,K] x Wt(bf16,[N][K]) + bias (+ resid)
// One wave per 16x(16*NG) output strip. A converted f32->bf16 in registers.
// ---------------------------------------------------------------------------
template <int NG>
__launch_bounds__(256)
__global__ void wmma_gemm_bf16(const float* __restrict__ A,
                               const __bf16* __restrict__ Bw,
                               const float* __restrict__ bias,
                               const float* __restrict__ resid,
                               float* __restrict__ C,
                               int Mrows, int N, int K) {
    const int lane  = threadIdx.x & 31;
    const int wave  = blockIdx.x * (blockDim.x >> 5) + (threadIdx.x >> 5);
    const int ngrps = N / (16 * NG);
    const int mtiles = Mrows >> 4;
    if (wave >= mtiles * ngrps) return;

    const int mt   = wave / ngrps;
    const int ng   = wave - mt * ngrps;
    const int m0   = mt << 4;
    const int n0   = ng * (16 * NG);
    const int half = lane >> 4;     // 0: lanes 0-15, 1: lanes 16-31
    const int l15  = lane & 15;

    const float* arow = A + (size_t)(m0 + l15) * K;   // A row = lane&15 for both halves

    v8f acc[NG] = {};

    for (int k0 = 0; k0 < K; k0 += 32) {
        // ---- A fragment: K in {kb..kb+15}, kb = k0 (+16 for upper half)
        const int kb = k0 + (half ? 16 : 0);
        float4 f0 = *reinterpret_cast<const float4*>(arow + kb);
        float4 f1 = *reinterpret_cast<const float4*>(arow + kb + 4);
        float4 f2 = *reinterpret_cast<const float4*>(arow + kb + 8);
        float4 f3 = *reinterpret_cast<const float4*>(arow + kb + 12);
        v16bf a;
        a[0]  = (__bf16)f0.x; a[1]  = (__bf16)f0.y; a[2]  = (__bf16)f0.z; a[3]  = (__bf16)f0.w;
        a[4]  = (__bf16)f1.x; a[5]  = (__bf16)f1.y; a[6]  = (__bf16)f1.z; a[7]  = (__bf16)f1.w;
        a[8]  = (__bf16)f2.x; a[9]  = (__bf16)f2.y; a[10] = (__bf16)f2.z; a[11] = (__bf16)f2.w;
        a[12] = (__bf16)f3.x; a[13] = (__bf16)f3.y; a[14] = (__bf16)f3.z; a[15] = (__bf16)f3.w;

        // ---- B fragments: lane -> column n0+g*16+l15, 16 consecutive K (upper half +16)
        const int kr = k0 + (half ? 16 : 0);
#pragma unroll
        for (int g = 0; g < NG; ++g) {
            const __bf16* bp = Bw + ((size_t)(n0 + g * 16 + l15) * K + kr);
            v16bf b = *reinterpret_cast<const v16bf*>(bp);
            acc[g] = __builtin_amdgcn_wmma_f32_16x16x32_bf16(
                false, a, false, b, (short)0, acc[g], false, false);
        }
    }

    // ---- Epilogue: C element (vgpr j, lane) = row m0+j+8*half, col n0+g*16+l15
#pragma unroll
    for (int g = 0; g < NG; ++g) {
        const int col = n0 + g * 16 + l15;
        const float bv = bias ? bias[col] : 0.0f;
#pragma unroll
        for (int j = 0; j < 8; ++j) {
            const int r = m0 + j + (half << 3);
            float o = acc[g][j] + bv;
            if (resid) o += resid[(size_t)r * N + col];
            C[(size_t)r * N + col] = o;
        }
    }
}

// ---------------------------------------------------------------------------
// Softmax + bilinear deformable sampling. One wave per (b, q, m); lane = d.
// Gathers from v (projected value, f32 [B*NV, 256], col = m*32+d), which is
// L2-resident (40 MB << 192 MB).
// ---------------------------------------------------------------------------
__device__ __forceinline__ float fetch_corner(const float* __restrict__ vl,
                                              int H, int W, int ix, int iy) {
    if (ix < 0 || ix >= W || iy < 0 || iy >= H) return 0.0f;
    return vl[(size_t)(iy * W + ix) * kE];
}

__launch_bounds__(256)
__global__ void msda_sample(const float* __restrict__ v,
                            const float* __restrict__ offb,
                            const float* __restrict__ attnb,
                            const float* __restrict__ refp,
                            float* __restrict__ samp) {
    const int  lane  = threadIdx.x & 31;
    const long wave  = (long)blockIdx.x * (blockDim.x >> 5) + (threadIdx.x >> 5);
    const long total = (long)kB * kLQ * kM;
    if (wave >= total) return;

    const int  m  = (int)(wave % kM);
    const long bq = wave / kM;
    const int  q  = (int)(bq % kLQ);
    const int  b  = (int)(bq / kLQ);

    // softmax over 16 logits (l*P + p)
    const float* lg = attnb + ((size_t)(b * kLQ + q) * (kM * kL * kP) + m * (kL * kP));
    float e[16];
    float mx = lg[0];
#pragma unroll
    for (int i = 1; i < 16; ++i) mx = fmaxf(mx, lg[i]);
    float s = 0.0f;
#pragma unroll
    for (int i = 0; i < 16; ++i) { e[i] = __expf(lg[i] - mx); s += e[i]; }
    const float inv = 1.0f / s;

    const float* rp = refp + (size_t)(b * kLQ + q) * (kL * 2);
    const float* of = offb + ((size_t)(b * kLQ + q) * (kM * kL * kP * 2) + m * (kL * kP * 2));
    const float* vb = v + (size_t)b * kNV * kE + m * kD + lane;

    float acc = 0.0f;
#pragma unroll
    for (int l = 0; l < kL; ++l) {
        const int H = cLVL_H[l], W = cLVL_W[l];
        const float* vl = vb + (size_t)cLVL_S[l] * kE;
        const float rx = rp[l * 2 + 0], ry = rp[l * 2 + 1];
        const float invW = 1.0f / (float)W, invH = 1.0f / (float)H;
#pragma unroll
        for (int p = 0; p < kP; ++p) {
            const float ox = of[(l * kP + p) * 2 + 0];
            const float oy = of[(l * kP + p) * 2 + 1];
            const float lx = rx + ox * invW;
            const float ly = ry + oy * invH;
            const float x = lx * (float)W - 0.5f;
            const float y = ly * (float)H - 0.5f;
            const float xf = floorf(x), yf = floorf(y);
            const float fx = x - xf, fy = y - yf;
            const int x0 = (int)xf, y0 = (int)yf;
            const float c00 = fetch_corner(vl, H, W, x0,     y0);
            const float c10 = fetch_corner(vl, H, W, x0 + 1, y0);
            const float c01 = fetch_corner(vl, H, W, x0,     y0 + 1);
            const float c11 = fetch_corner(vl, H, W, x0 + 1, y0 + 1);
            const float bil = (1.0f - fx) * (1.0f - fy) * c00 + fx * (1.0f - fy) * c10 +
                              (1.0f - fx) * fy * c01 + fx * fy * c11;
            acc += e[l * kP + p] * inv * bil;
        }
    }
    samp[(size_t)(b * kLQ + q) * kE + m * kD + lane] = acc;
}

// ---------------------------------------------------------------------------
// Host-side launch sequence (all on `stream`; graph-capture safe).
// ---------------------------------------------------------------------------
extern "C" void kernel_launch(void* const* d_in, const int* in_sizes, int n_in,
                              void* d_out, int out_size, void* d_ws, size_t ws_size,
                              hipStream_t stream) {
    const float* query = (const float*)d_in[0];
    const float* value = (const float*)d_in[1];
    const float* refp  = (const float*)d_in[2];
    // d_in[3] spatial_shapes: compile-time constant, unused
    const float* Wv    = (const float*)d_in[4];
    const float* bv    = (const float*)d_in[5];
    const float* Woff  = (const float*)d_in[6];
    const float* boff  = (const float*)d_in[7];
    const float* Wattn = (const float*)d_in[8];
    const float* battn = (const float*)d_in[9];
    const float* Wout  = (const float*)d_in[10];
    const float* bout  = (const float*)d_in[11];
    float* out = (float*)d_out;

    // Workspace carve-up (~100.6 MB)
    char* ws = (char*)d_ws;
    size_t pos = 0;
    auto carve = [&](size_t bytes) {
        void* p = ws + pos;
        pos += (bytes + 255) & ~(size_t)255;
        return p;
    };
    float*  v_f32  = (float*) carve((size_t)kROWS * kE * sizeof(float));          // 40 MB
    float*  offbuf = (float*) carve((size_t)kROWS * kE * sizeof(float));          // 40 MB
    float*  attnbf = (float*) carve((size_t)kROWS * (kM * kL * kP) * sizeof(float)); // 20 MB
    __bf16* Wv_t   = (__bf16*)carve((size_t)kE * kE * sizeof(__bf16));
    __bf16* Woff_t = (__bf16*)carve((size_t)kE * kE * sizeof(__bf16));
    __bf16* Wattn_t= (__bf16*)carve((size_t)kE * (kM * kL * kP) * sizeof(__bf16));
    __bf16* Wout_t = (__bf16*)carve((size_t)kE * kE * sizeof(__bf16));
    (void)ws_size; (void)in_sizes; (void)n_in; (void)out_size;

    const int TB = 256;

    // 1) pack weights -> transposed bf16 (tiny, L2-resident thereafter)
    pack_w_bf16<<<(kE * kE + TB - 1) / TB, TB, 0, stream>>>(Wv, Wv_t, kE, kE);
    pack_w_bf16<<<(kE * kE + TB - 1) / TB, TB, 0, stream>>>(Woff, Woff_t, kE, kE);
    pack_w_bf16<<<(kE * 128 + TB - 1) / TB, TB, 0, stream>>>(Wattn, Wattn_t, 128, kE);
    pack_w_bf16<<<(kE * kE + TB - 1) / TB, TB, 0, stream>>>(Wout, Wout_t, kE, kE);

    const int mtiles = kROWS >> 4;            // 2445
    auto gemm_blocks = [&](int N) {
        int waves = mtiles * (N / 64);        // NG=4 -> 64 cols per wave
        return (waves + 7) / 8;               // 8 waves per 256-thread block
    };

    // 2) v = value @ Wv + bv            [39120 x 256]
    wmma_gemm_bf16<4><<<gemm_blocks(256), TB, 0, stream>>>(
        value, Wv_t, bv, nullptr, v_f32, kROWS, 256, kE);
    // 3) offsets = query @ Woff + boff  [39120 x 256]
    wmma_gemm_bf16<4><<<gemm_blocks(256), TB, 0, stream>>>(
        query, Woff_t, boff, nullptr, offbuf, kROWS, 256, kE);
    // 4) attn logits = query @ Wattn + battn  [39120 x 128]
    wmma_gemm_bf16<4><<<gemm_blocks(128), TB, 0, stream>>>(
        query, Wattn_t, battn, nullptr, attnbf, kROWS, 128, kE);

    // 5) softmax + bilinear sampling -> writes into d_out (used as f32 scratch)
    {
        long waves = (long)kB * kLQ * kM;     // 312960
        int blocks = (int)((waves + 7) / 8);
        msda_sample<<<blocks, TB, 0, stream>>>(v_f32, offbuf, attnbf, refp, out);
    }

    // 6) out = samp @ Wout + bout + query  (in-place: each wave reads exactly
    //    the 16 rows it writes, finishing all reads before the epilogue)
    wmma_gemm_bf16<4><<<gemm_blocks(256), TB, 0, stream>>>(
        out, Wout_t, bout, query, out, kROWS, 256, kE);
}
